// he_gat_52338471469729
// MI455X (gfx1250) — compile-verified, matrix operand order checked by reading, not run on hardware
//
#include <hip/hip_runtime.h>
#include <hip/hip_bf16.h>

typedef __attribute__((ext_vector_type(16))) _Float16 v16h;
typedef __attribute__((ext_vector_type(8)))  _Float16 v8h;
typedef __attribute__((ext_vector_type(8)))  float    v8f;

constexpr int B_ = 32;
constexpr int N_ = 1024;
constexpr int H_ = 256;

// ---------------- WMMA fragment loaders ----------------
// A-matrix 16x32 f16 frag from row-major f16 source (ISA 7.12.2 layout):
// lanes 0-15: row m=lane, K 0..7 in v0..3, K 16..23 in v4..7; lanes 16-31: +8.
__device__ __forceinline__ v16h load_a_f16(const _Float16* src, int ld) {
    int lane = threadIdx.x & 31;
    int m = lane & 15, g = lane >> 4;
    const _Float16* p = src + (size_t)m * ld + g * 8;
    v8h lo = *(const v8h*)(p);        // K = g*8 .. g*8+7
    v8h hi = *(const v8h*)(p + 16);   // K = 16+g*8 ..
    v16h a;
#pragma unroll
    for (int i = 0; i < 8; ++i) { a[i] = lo[i]; a[8 + i] = hi[i]; }
    return a;
}

// Same A layout from fp32 row-major source, converting to f16.
__device__ __forceinline__ v16h load_a_f32(const float* src, int ld) {
    int lane = threadIdx.x & 31;
    int m = lane & 15, g = lane >> 4;
    const float* p = src + (size_t)m * ld + g * 8;
    v16h a;
#pragma unroll
    for (int i = 0; i < 8; ++i) {
        a[i]     = (_Float16)p[i];
        a[8 + i] = (_Float16)p[16 + i];
    }
    return a;
}

// B frag from pre-swizzled storage: tile = [lane 0..31][16 halves] = 1KB,
// one contiguous 32B load per lane.
__device__ __forceinline__ v16h load_b_sw(const _Float16* tile) {
    int lane = threadIdx.x & 31;
    return *(const v16h*)(tile + (size_t)lane * 16);
}

#define WMMA_F16(Afrag, Bfrag, Cacc) \
    __builtin_amdgcn_wmma_f32_16x16x32_f16(false, (Afrag), false, (Bfrag), (short)0, (Cacc), false, false)

// ---------------- K0: pre-swizzle W_i / W_c into B-frag-native f16 layout ----------
// Layout: [kb (H/32=8)][et (H/16=16)][lane (32)][pos (16)] halves.
__global__ __launch_bounds__(256)
void gat_prep_w(const float* __restrict__ Wi, const float* __restrict__ Wc,
                _Float16* __restrict__ wisw, _Float16* __restrict__ wcsw) {
    int idx  = blockIdx.x * 256 + threadIdx.x;   // 0 .. 65535 (one per half)
    int pos  = idx & 15;
    int lane = (idx >> 4) & 31;
    int et   = (idx >> 9) & 15;
    int kb   = idx >> 13;                        // 0..7
    int n = lane & 15, g = lane >> 4;
    int q = pos >> 1;
    int k = (q < 4 ? 2 * q : 16 + 2 * (q - 4)) + 8 * g + (pos & 1) + kb * 32;
    int e = et * 16 + n;
    wisw[idx] = (_Float16)Wi[k * H_ + e];
    wcsw[idx] = (_Float16)Wc[k * H_ + e];
}

// ================= K1: Wh = select(inp@W_i, inp@W_c) =================
struct K1Frags { v16h af; v16h bi[4]; v16h bc[4]; };

__device__ __forceinline__ void k1_load(K1Frags& f, const float* Abase,
                                        const _Float16* wisw, const _Float16* wcsw,
                                        int k, int eb) {
    f.af = load_a_f32(Abase + k, H_);
    int kb = k >> 5;
#pragma unroll
    for (int t = 0; t < 4; ++t) {
        f.bi[t] = load_b_sw(wisw + ((size_t)kb * 16 + eb + t) * 512);
        f.bc[t] = load_b_sw(wcsw + ((size_t)kb * 16 + eb + t) * 512);
    }
}

__device__ __forceinline__ void k1_compute(const K1Frags& f, v8f acc_i[4], v8f acc_c[4]) {
#pragma unroll
    for (int t = 0; t < 4; ++t) {
        acc_i[t] = WMMA_F16(f.af, f.bi[t], acc_i[t]);
        acc_c[t] = WMMA_F16(f.af, f.bc[t], acc_c[t]);
    }
}

// grid.x = B * (N/32); block = 256 (8 waves). WG tile 32(M) x 256(E); wave 16x64.
// Ping-pong pipelined (no rotate copies). Epilogue writes row-major f16 Wh
// (for K1b) AND swizzled Wh (for K3).
__global__ __launch_bounds__(256)
void gat_gemm_wh(const float* __restrict__ inp, const _Float16* __restrict__ wisw,
                 const _Float16* __restrict__ wcsw, const int* __restrict__ l,
                 _Float16* __restrict__ whrow, _Float16* __restrict__ whsw) {
    int bid  = blockIdx.x;
    int b    = bid >> 5;
    int tM   = (bid & 31) * 32;
    int wave = threadIdx.x >> 5;
    int lane = threadIdx.x & 31;
    int wm   = (wave >> 2) * 16;      // 0 / 16
    int wn   = (wave & 3) * 64;       // 0..192
    int eb   = wn >> 4;

    const float* Abase = inp + (size_t)b * N_ * H_ + (size_t)(tM + wm) * H_;

    v8f acc_i[4] = {{0}}, acc_c[4] = {{0}};
    K1Frags fa, fb;
    k1_load(fa, Abase, wisw, wcsw, 0, eb);
    for (int k = 0; k < H_ - 64; k += 64) {
        k1_load(fb, Abase, wisw, wcsw, k + 32, eb);
        k1_compute(fa, acc_i, acc_c);
        k1_load(fa, Abase, wisw, wcsw, k + 64, eb);
        k1_compute(fb, acc_i, acc_c);
    }
    k1_load(fb, Abase, wisw, wcsw, H_ - 32, eb);
    k1_compute(fa, acc_i, acc_c);
    k1_compute(fb, acc_i, acc_c);

    int l0 = l[2 * b], l1 = l[2 * b + 1];
    int nC = lane & 15, goff = (lane >> 4) * 8;
    int kb_out = tM >> 5;             // node-dim K-block for K3's B layout
    int hi     = wm >> 4;             // pos offset: 8*hi + r
#pragma unroll
    for (int t = 0; t < 4; ++t) {
        v8h h;
#pragma unroll
        for (int r = 0; r < 8; ++r) {
            int m = tM + wm + r + goff;
            bool sel = (m >= l0) & (m < l1);
            float v = sel ? acc_i[t][r] : acc_c[t][r];
            h[r] = (_Float16)v;
            whrow[(size_t)b * N_ * H_ + (size_t)m * H_ + wn + t * 16 + nC] = h[r];
        }
        int et = eb + t;
        // swizzled: [b][kb=32][et=16][lane=32][pos=16]; same lane, contiguous pos.
        *(v8h*)(whsw + ((((size_t)b * 32 + kb_out) * 16 + et) * 32 + lane) * 16 + 8 * hi) = h;
    }
}

// ---------------- K1b: Wh1/Wh2 = Wh @ a[:H] / a[H:]  (wave per row) ----------------
__global__ __launch_bounds__(256)
void gat_attvec(const _Float16* __restrict__ whrow, const float* __restrict__ a,
                float* __restrict__ wh1, float* __restrict__ wh2) {
    int row  = blockIdx.x * 8 + (threadIdx.x >> 5);   // [0, B*N)
    int lane = threadIdx.x & 31;
    const _Float16* p = whrow + (size_t)row * H_ + lane * 8;
    v8h v = __builtin_nontemporal_load((const v8h*)p);   // read-once stream
    float s1 = 0.f, s2 = 0.f;
#pragma unroll
    for (int i = 0; i < 8; ++i) {
        float x = (float)v[i];
        s1 += x * a[lane * 8 + i];
        s2 += x * a[H_ + lane * 8 + i];
    }
#pragma unroll
    for (int off = 16; off > 0; off >>= 1) {
        s1 += __shfl_xor(s1, off, 32);
        s2 += __shfl_xor(s2, off, 32);
    }
    if (lane == 0) { wh1[row] = s1; wh2[row] = s2; }
}

// ---------------- K2: masked leaky-relu + row softmax -> att (f16) ----------------
// Adjacency is a 128MB read-once stream: NT loads keep it from evicting the
// reused att/whsw tensors out of the 192MB L2 before K3 consumes them.
__global__ __launch_bounds__(256)
void gat_softmax(const float* __restrict__ A, const float* __restrict__ wh1,
                 const float* __restrict__ wh2, _Float16* __restrict__ att) {
    int row = blockIdx.x;             // b*N + i
    int b   = row >> 10;
    int i   = row & (N_ - 1);
    int tid = threadIdx.x;
    const float* Arow = A + (size_t)row * N_;
    float h1 = wh1[row];

    float e[4];
    float mx = -3.0e38f;
#pragma unroll
    for (int t = 0; t < 4; ++t) {
        int j = tid + t * 256;
        float a_ = __builtin_nontemporal_load(Arow + j) + ((j == i) ? 1.0f : 0.0f);
        float x;
        if (a_ > 0.0f) {
            float s = (h1 + wh2[b * N_ + j]) * a_;
            x = (s > 0.0f) ? s : 0.2f * s;               // leaky relu, alpha=0.2
        } else {
            x = -1.0e12f;                                 // NEG_INF mask
        }
        e[t] = x;
        mx = fmaxf(mx, x);
    }

    __shared__ float red[256];
    red[tid] = mx; __syncthreads();
#pragma unroll
    for (int s = 128; s > 0; s >>= 1) {
        if (tid < s) red[tid] = fmaxf(red[tid], red[tid + s]);
        __syncthreads();
    }
    mx = red[0]; __syncthreads();

    float sum = 0.f;
#pragma unroll
    for (int t = 0; t < 4; ++t) { e[t] = __expf(e[t] - mx); sum += e[t]; }
    red[tid] = sum; __syncthreads();
#pragma unroll
    for (int s = 128; s > 0; s >>= 1) {
        if (tid < s) red[tid] += red[tid + s];
        __syncthreads();
    }
    float inv = 1.0f / red[0];
#pragma unroll
    for (int t = 0; t < 4; ++t)
        att[(size_t)row * N_ + tid + t * 256] = (_Float16)(e[t] * inv);
}

// ================= K3: out = elu(att @ Wh) =================
struct K3Frags { v16h af0, af1; v16h bf[4]; };

__device__ __forceinline__ void k3_load(K3Frags& f, const _Float16* A0,
                                        const _Float16* A1, const _Float16* Bsw,
                                        int k, int eb) {
    f.af0 = load_a_f16(A0 + k, N_);
    f.af1 = load_a_f16(A1 + k, N_);
    int kb = k >> 5;
#pragma unroll
    for (int t = 0; t < 4; ++t)
        f.bf[t] = load_b_sw(Bsw + ((size_t)kb * 16 + eb + t) * 512);
}

__device__ __forceinline__ void k3_compute(const K3Frags& f, v8f acc0[4], v8f acc1[4]) {
#pragma unroll
    for (int t = 0; t < 4; ++t) {
        acc0[t] = WMMA_F16(f.af0, f.bf[t], acc0[t]);
        acc1[t] = WMMA_F16(f.af1, f.bf[t], acc1[t]);
    }
}

// grid = B * (N/64) = 512; block = 256 (8 waves). WG tile 64(M) x 256(E).
// Wave tile 32x64: 2 A-frags share 4 B-frags -> 8 WMMA per 12 b128 loads.
// Ping-pong pipelined; output stores are non-temporal (never re-read).
__global__ __launch_bounds__(256)
void gat_gemm_out(const _Float16* __restrict__ att, const _Float16* __restrict__ whsw,
                  float* __restrict__ out) {
    int bid  = blockIdx.x;
    int b    = bid >> 4;              // 16 M-tiles of 64 per batch
    int tM   = (bid & 15) * 64;
    int wave = threadIdx.x >> 5;
    int lane = threadIdx.x & 31;
    int wm   = (wave >> 2) * 32;      // 0 / 32
    int wn   = (wave & 3) * 64;       // 0..192
    int eb   = wn >> 4;

    const _Float16* A0  = att + (size_t)b * N_ * N_ + (size_t)(tM + wm) * N_;
    const _Float16* A1  = A0 + (size_t)16 * N_;
    const _Float16* Bsw = whsw + (size_t)b * (32 * 16 * 512);

    v8f acc0[4] = {{0}}, acc1[4] = {{0}};
    K3Frags fa, fb;
    k3_load(fa, A0, A1, Bsw, 0, eb);
    for (int k = 0; k < N_ - 64; k += 64) {
        __builtin_prefetch(A0 + k + 256, 0, 1);
        __builtin_prefetch(A1 + k + 256, 0, 1);
        k3_load(fb, A0, A1, Bsw, k + 32, eb);
        k3_compute(fa, acc0, acc1);
        k3_load(fa, A0, A1, Bsw, k + 64, eb);
        k3_compute(fb, acc0, acc1);
    }
    k3_load(fb, A0, A1, Bsw, N_ - 32, eb);
    k3_compute(fa, acc0, acc1);
    k3_compute(fb, acc0, acc1);

    int nC = lane & 15, goff = (lane >> 4) * 8;
    float* outb = out + (size_t)b * N_ * H_;
#pragma unroll
    for (int t = 0; t < 4; ++t)
#pragma unroll
        for (int r = 0; r < 8; ++r) {
            int m = tM + wm + r + goff;
            float v = acc0[t][r];
            v = (v > 0.0f) ? v : (__expf(v) - 1.0f);      // elu
            __builtin_nontemporal_store(v, outb + (size_t)m * H_ + wn + t * 16 + nC);
        }
#pragma unroll
    for (int t = 0; t < 4; ++t)
#pragma unroll
        for (int r = 0; r < 8; ++r) {
            int m = tM + wm + 16 + r + goff;
            float v = acc1[t][r];
            v = (v > 0.0f) ? v : (__expf(v) - 1.0f);      // elu
            __builtin_nontemporal_store(v, outb + (size_t)m * H_ + wn + t * 16 + nC);
        }
}

// ---------------- launcher ----------------
extern "C" void kernel_launch(void* const* d_in, const int* in_sizes, int n_in,
                              void* d_out, int out_size, void* d_ws, size_t ws_size,
                              hipStream_t stream) {
    const float* inp = (const float*)d_in[0];   // [B,N,H]
    const float* A   = (const float*)d_in[1];   // [B,N,N]
    const int*   l   = (const int*)  d_in[2];   // [B,2]
    const float* Wi  = (const float*)d_in[3];   // [H,H]
    const float* Wc  = (const float*)d_in[4];   // [H,H]
    const float* a   = (const float*)d_in[5];   // [2H,1]
    float* out = (float*)d_out;                 // [B,N,H]

    char* ws = (char*)d_ws;
    size_t off = 0;
    _Float16* whsw  = (_Float16*)(ws + off); off += (size_t)B_ * N_ * H_ * sizeof(_Float16); // 16 MB
    _Float16* whrow = (_Float16*)(ws + off); off += (size_t)B_ * N_ * H_ * sizeof(_Float16); // 16 MB
    _Float16* att   = (_Float16*)(ws + off); off += (size_t)B_ * N_ * N_ * sizeof(_Float16); // 64 MB
    float* wh1 = (float*)(ws + off); off += (size_t)B_ * N_ * sizeof(float);
    float* wh2 = (float*)(ws + off); off += (size_t)B_ * N_ * sizeof(float);
    _Float16* wisw = (_Float16*)(ws + off); off += (size_t)H_ * H_ * sizeof(_Float16);       // 128 KB
    _Float16* wcsw = (_Float16*)(ws + off); off += (size_t)H_ * H_ * sizeof(_Float16);       // 128 KB
    (void)ws_size; (void)in_sizes; (void)n_in; (void)out_size;

    dim3 blk(256);
    // K0: pre-swizzle weights into B-frag layout
    gat_prep_w<<<dim3((H_ * H_) / 256), blk, 0, stream>>>(Wi, Wc, wisw, wcsw);
    // K1: dual GEMM + row select
    gat_gemm_wh<<<dim3(B_ * (N_ / 32)), blk, 0, stream>>>(inp, wisw, wcsw, l, whrow, whsw);
    // K1b: attention vectors
    gat_attvec<<<dim3((B_ * N_) / 8), blk, 0, stream>>>(whrow, a, wh1, wh2);
    // K2: masked leaky-relu + softmax
    gat_softmax<<<dim3(B_ * N_), blk, 0, stream>>>(A, wh1, wh2, att);
    // K3: att @ Wh + ELU
    gat_gemm_out<<<dim3(B_ * (N_ / 64)), blk, 0, stream>>>(att, whsw, out);
}
